// Attention_60593398612481
// MI455X (gfx1250) — compile-verified, matrix operand order checked by reading, not run on hardware
//
#include <hip/hip_runtime.h>
#include <hip/hip_bf16.h>

// ---------------------------------------------------------------------------
// Attention block for MI455X (gfx1250, wave32, WMMA + TDM).
//   xq = x@wq ; xk = x@wk ; xv = x@wv      (WMMA bf16 GEMM, f32->bf16 staging)
//   RoPE(xq), RoPE(xk)
//   ctx = flash-attention(xq, xk, xv)      (WMMA bf16, TDM tile staging)
//   out = ctx @ wo                         (WMMA bf16 GEMM, f32 out)
// ---------------------------------------------------------------------------

#define DIMV   4096
#define SEQ    2048
#define BSZV   2
#define NH     32
#define NKV    8
#define HD     128

typedef __attribute__((ext_vector_type(16))) __bf16 v16bf;
typedef __attribute__((ext_vector_type(8)))  __bf16 v8bf;
typedef __attribute__((ext_vector_type(4)))  __bf16 v4bf;
typedef __attribute__((ext_vector_type(8)))  float  v8f;
typedef __attribute__((ext_vector_type(4)))  float  v4f;
typedef __attribute__((ext_vector_type(4)))  unsigned v4u;
typedef __attribute__((ext_vector_type(8)))  int    v8i;
typedef __attribute__((ext_vector_type(4)))  int    v4i;

#if __has_builtin(__builtin_amdgcn_tensor_load_to_lds)
#define HAVE_TDM 1
#endif

__device__ __forceinline__ __bf16 f2bf(float f) {
  union { float f; unsigned u; } x; x.f = f;
  unsigned r = x.u + 0x7FFFu + ((x.u >> 16) & 1u);     // round-to-nearest-even
  unsigned short h = (unsigned short)(r >> 16);
  union { unsigned short s; __bf16 b; } y; y.s = h;
  return y.b;
}
__device__ __forceinline__ float bf2f(__bf16 b) {
  union { unsigned short s; __bf16 b; } y; y.b = b;
  union { unsigned u; float f; } x; x.u = ((unsigned)y.s) << 16;
  return x.f;
}

// Load a 16x32 bf16 fragment (A or B operand) from LDS, row-major, given row
// stride in elements.  Matches CDNA5 16-bit operand layout: lanes 0-15 hold
// K=0..7 and 16..23, lanes 16-31 hold K=8..15 and 24..31 for row (lane&15).
__device__ __forceinline__ v16bf load_frag(const __bf16* p0, int stride) {
  const int lane = threadIdx.x & 31;
  const __bf16* p = p0 + (lane & 15) * stride + ((lane >> 4) << 3);
  union { v16bf v; v8bf h[2]; } u;
  u.h[0] = *(const v8bf*)p;          // K block 0..7 (or 8..15)
  u.h[1] = *(const v8bf*)(p + 16);   // K block 16..23 (or 24..31)
  return u.v;
}

__device__ __forceinline__ v8f wmma_bf16(v16bf a, v16bf b, v8f c) {
  // (neg_a, A, neg_b, B, c_mod, C, reuse_a, reuse_b)
  return __builtin_amdgcn_wmma_f32_16x16x32_bf16(false, a, false, b,
                                                 (short)0, c, false, false);
}

#ifdef HAVE_TDM
// Issue one TDM 2D tile load (bf16 elements) global -> LDS.
//   lds_off        : LDS byte offset of tile destination
//   gptr           : global address of tile start (element [0,0])
//   tile_d0/d1     : tile width (contiguous elems) / height (rows)
//   row_stride     : tensor dim0 stride in elements (between rows)
//   pad_int/pad_amt: D# pad codes (pad_amt==0 -> no padding); dest row pitch
//                    becomes tile_d0*2 + 4*(pad_amt+1) bytes.
__device__ __forceinline__ void tdm_load_2d_bf16(
    unsigned lds_off, const void* gptr, unsigned tile_d0, unsigned tile_d1,
    unsigned row_stride, unsigned pad_int, unsigned pad_amt) {
  unsigned long long ga = (unsigned long long)(uintptr_t)gptr;
  v4u g0;
  g0[0] = 1u;                                        // count=1, user descriptor
  g0[1] = lds_off;                                   // lds_addr
  g0[2] = (unsigned)(ga & 0xFFFFFFFFu);              // global_addr[31:0]
  g0[3] = (unsigned)((ga >> 32) & 0x01FFFFFFu)       // global_addr[56:32]
          | (2u << 30);                              // type=2 ("image")
  unsigned flags = (1u << 16);                       // data_size=1 (2 bytes)
  if (pad_amt)
    flags |= (1u << 20) | (pad_int << 22) | (pad_amt << 25);  // pad_enable
  v8i g1;
  g1[0] = (int)flags;                                // mask=0 | size | pad
  g1[1] = (int)((tile_d0 & 0xFFFFu) << 16);          // tensor_dim0[15:0]
  g1[2] = (int)((tile_d0 >> 16) | ((tile_d1 & 0xFFFFu) << 16));   // dim0 hi|dim1 lo
  g1[3] = (int)((tile_d1 >> 16) | ((tile_d0 & 0xFFFFu) << 16));   // dim1 hi|tile_dim0
  g1[4] = (int)(tile_d1 & 0xFFFFu);                  // tile_dim1 | tile_dim2=0
  g1[5] = (int)row_stride;                           // tensor_dim0_stride[31:0]
  g1[6] = 0;                                         // stride hi | dim1_stride lo
  g1[7] = 0;                                         // dim1_stride hi (unused 2D)
  v4i gz4 = {0, 0, 0, 0};                            // groups 2/3 unused (2D)
  v8i gz8 = {0, 0, 0, 0, 0, 0, 0, 0};
  // 6-arg form (clang-23 / therock-10.0 headers):
  //   (v4u g0, v8i g1, v4i g2, v4i g3, v8i g4, i32 cpol)
  __builtin_amdgcn_tensor_load_to_lds(g0, g1, gz4, gz4, gz8, 0);
}
#endif

__device__ __forceinline__ void wait_tensorcnt0() {
#if __has_builtin(__builtin_amdgcn_s_wait_tensorcnt)
  __builtin_amdgcn_s_wait_tensorcnt(0);
#else
  asm volatile("s_wait_tensorcnt 0" ::: "memory");
#endif
}

// ---------------------------------------------------------------------------
// GEMM: C[M,N] = A[M,K] * B[K,N].  A is f32 or bf16 (template), B is f32
// (converted to bf16 while staging), C is bf16 or f32 (template).
// Block = 256 threads = 8 waves, tile 128x128, K-step 32.
// ---------------------------------------------------------------------------
template <bool ABF16, bool OUTF32>
__global__ __launch_bounds__(256) void gemm_wmma_kernel(
    const void* __restrict__ Aptr, const float* __restrict__ B,
    void* __restrict__ Cptr, int M, int N, int K) {
  constexpr int LDT = 40;                       // 80B row stride (16B aligned)
  __shared__ __align__(16) __bf16 As[128 * LDT];
  __shared__ __align__(16) __bf16 Bs[128 * LDT]; // transposed: Bs[n][k]

  const int tid  = threadIdx.x;
  const int lane = tid & 31;
  const int wid  = tid >> 5;
  const int wm   = (wid >> 2) * 64;             // wave row offset in tile
  const int wn   = (wid & 3) * 32;              // wave col offset in tile
  const int rowBase = blockIdx.y * 128;
  const int colBase = blockIdx.x * 128;

  v8f acc[4][2];
  const v8f vzero = {0.f, 0.f, 0.f, 0.f, 0.f, 0.f, 0.f, 0.f};
#pragma unroll
  for (int mt = 0; mt < 4; ++mt)
#pragma unroll
    for (int nt = 0; nt < 2; ++nt) acc[mt][nt] = vzero;

  for (int k0 = 0; k0 < K; k0 += 32) {
    // ---- stage A tile (128 x 32) ----
    if (ABF16) {
      const __bf16* A = (const __bf16*)Aptr;
#pragma unroll
      for (int i = 0; i < 2; ++i) {             // 512 chunks of 8 bf16
        int c = i * 256 + tid;
        int r = c >> 2, c8 = (c & 3) * 8;
        *(v8bf*)&As[r * LDT + c8] =
            *(const v8bf*)(A + (size_t)(rowBase + r) * K + k0 + c8);
      }
    } else {
      const float* A = (const float*)Aptr;
#pragma unroll
      for (int i = 0; i < 4; ++i) {             // 1024 chunks of 4 f32
        int c = i * 256 + tid;
        int r = c >> 3, c4 = (c & 7) * 4;
        v4f f = *(const v4f*)(A + (size_t)(rowBase + r) * K + k0 + c4);
        v4bf h;
        h[0] = f2bf(f[0]); h[1] = f2bf(f[1]); h[2] = f2bf(f[2]); h[3] = f2bf(f[3]);
        *(v4bf*)&As[r * LDT + c4] = h;
      }
    }
    // ---- stage B tile (32 x 128), transposed into Bs[n][k] ----
#pragma unroll
    for (int i = 0; i < 4; ++i) {               // 1024 chunks of 4 f32
      int c = i * 256 + tid;
      int kk = c >> 5, n4 = (c & 31) * 4;
      v4f f = *(const v4f*)(B + (size_t)(k0 + kk) * N + colBase + n4);
#pragma unroll
      for (int j = 0; j < 4; ++j) Bs[(n4 + j) * LDT + kk] = f2bf(f[j]);
    }
    if (k0 + 32 < K) {
      __builtin_prefetch(B + (size_t)(k0 + 32) * N + colBase + lane * 4, 0, 1);
    }
    __syncthreads();

    v16bf af[4];
#pragma unroll
    for (int mt = 0; mt < 4; ++mt)
      af[mt] = load_frag(&As[(wm + mt * 16) * LDT], LDT);
#pragma unroll
    for (int nt = 0; nt < 2; ++nt) {
      v16bf bf_ = load_frag(&Bs[(wn + nt * 16) * LDT], LDT);
#pragma unroll
      for (int mt = 0; mt < 4; ++mt)
        acc[mt][nt] = wmma_bf16(af[mt], bf_, acc[mt][nt]);
    }
    __syncthreads();
  }

  // ---- epilogue: C layout lane(l) vgpr(r) -> (M = r + 8*(l>>4), N = l&15) ----
  const int hi8 = (lane >> 4) << 3;
  const int cN  = lane & 15;
#pragma unroll
  for (int mt = 0; mt < 4; ++mt)
#pragma unroll
    for (int nt = 0; nt < 2; ++nt)
#pragma unroll
      for (int r = 0; r < 8; ++r) {
        int row = rowBase + wm + mt * 16 + r + hi8;
        int col = colBase + wn + nt * 16 + cN;
        if (OUTF32)
          ((float*)Cptr)[(size_t)row * N + col] = acc[mt][nt][r];
        else
          ((__bf16*)Cptr)[(size_t)row * N + col] = f2bf(acc[mt][nt][r]);
      }
}

// ---------------------------------------------------------------------------
// RoPE (interleaved even/odd pairs), in place on bf16 [B, S, nHeads, HD].
// ---------------------------------------------------------------------------
__global__ __launch_bounds__(256) void rope_kernel(
    __bf16* __restrict__ t, const float* __restrict__ cosT,
    const float* __restrict__ sinT, int nHeads) {
  size_t idx = (size_t)blockIdx.x * 256 + threadIdx.x;  // one rotation pair
  int d = (int)(idx & 63);
  size_t tmp = idx >> 6;
  int h = (int)(tmp % nHeads);
  tmp /= nHeads;                       // tmp = b*SEQ + s
  int s = (int)(tmp & (SEQ - 1));
  size_t base = (tmp * nHeads + h) * HD + (size_t)d * 2;
  float c = cosT[s * 64 + d], sn = sinT[s * 64 + d];
  float e = bf2f(t[base]), o = bf2f(t[base + 1]);
  t[base]     = f2bf(e * c - o * sn);
  t[base + 1] = f2bf(e * sn + o * c);
}

// ---------------------------------------------------------------------------
// Flash attention (causal, GQA): one block per (batch*head, 128 q rows).
// Q tile and K chunks are DMA'd into LDS by the Tensor Data Mover (wave 0
// issues descriptors; TDM padding keeps the 136-element LDS row pitch).
// V is staged manually (transposed).  Each wave owns 16 q rows; online
// softmax with 16-lane shuffles; P staged through per-wave LDS to reshape
// C-layout -> A-layout.
// ---------------------------------------------------------------------------
__global__ __launch_bounds__(256) void attn_kernel(
    const __bf16* __restrict__ xq, const __bf16* __restrict__ xk,
    const __bf16* __restrict__ xv, __bf16* __restrict__ ctx) {
  constexpr int LQ = 136;  // row stride for Qs/Ks (128 + 8 pad, 16B aligned)
  constexpr int LV = 72;   // row stride for Vt/Ps (64 + 8 pad, 16B aligned)
  // TDM pad codes for a 256B row + 16B pad: interval 5 (=32 x 8B), amount 3 (=4 DW)
  constexpr unsigned PAD_INT = 5, PAD_AMT = 3;
  __shared__ __align__(16) __bf16 Qs[128 * LQ];
  __shared__ __align__(16) __bf16 Ks[64 * LQ];
  __shared__ __align__(16) __bf16 Vt[128 * LV];      // V transposed: [d][kv]
  __shared__ __align__(16) __bf16 Ps[8][16 * LV];    // per-wave P tile

  const int tid = threadIdx.x, lane = tid & 31, wid = tid >> 5;
  const int b = blockIdx.x >> 5, h = blockIdx.x & 31;
  const int kvh = h >> 2;                            // GQA: 4 q heads / kv head
  const int qBase = blockIdx.y * 128;
  const float scale = 0.08838834764831845f;          // 1/sqrt(128)
  const int hi8 = (lane >> 4) << 3;
  const int cN  = lane & 15;

  // ---- stage Q tile 128 x 128 ----
#ifdef HAVE_TDM
  if (wid == 0) {
    tdm_load_2d_bf16((unsigned)(uintptr_t)&Qs[0],
                     xq + ((size_t)(b * SEQ + qBase) * NH + h) * HD,
                     /*tile_d0=*/HD, /*tile_d1=*/128,
                     /*row_stride=*/NH * HD, PAD_INT, PAD_AMT);
  }
#else
#pragma unroll
  for (int i = 0; i < 8; ++i) {                      // 2048 chunks of 8 bf16
    int c = i * 256 + tid;
    int r = c >> 4, c8 = (c & 15) * 8;
    *(v8bf*)&Qs[r * LQ + c8] = *(const v8bf*)(
        xq + ((size_t)(b * SEQ + qBase + r) * NH + h) * HD + c8);
  }
#endif

  v8f O[8];
  const v8f vzero = {0.f, 0.f, 0.f, 0.f, 0.f, 0.f, 0.f, 0.f};
#pragma unroll
  for (int dt = 0; dt < 8; ++dt) O[dt] = vzero;
  float mrow[8], lrow[8];
#pragma unroll
  for (int r = 0; r < 8; ++r) { mrow[r] = -INFINITY; lrow[r] = 0.f; }

  const int nChunks = blockIdx.y * 2 + 2;            // causal: kv <= qBase+127
  for (int cidx = 0; cidx < nChunks; ++cidx) {
    const int kvBase = cidx * 64;
    // ---- stage K chunk 64 x 128 ----
#ifdef HAVE_TDM
    if (wid == 0) {
      tdm_load_2d_bf16((unsigned)(uintptr_t)&Ks[0],
                       xk + ((size_t)(b * SEQ + kvBase) * NKV + kvh) * HD,
                       /*tile_d0=*/HD, /*tile_d1=*/64,
                       /*row_stride=*/NKV * HD, PAD_INT, PAD_AMT);
    }
#else
#pragma unroll
    for (int i = 0; i < 4; ++i) {                    // 1024 chunks of 8 bf16
      int c = i * 256 + tid;
      int r = c >> 4, c8 = (c & 15) * 8;
      *(v8bf*)&Ks[r * LQ + c8] = *(const v8bf*)(
          xk + ((size_t)(b * SEQ + kvBase + r) * NKV + kvh) * HD + c8);
    }
#endif
    // ---- stage V chunk transposed: Vt[d][kv] (manual; TDM can't transpose) --
#pragma unroll
    for (int i = 0; i < 4; ++i) {
      int c = i * 256 + tid;
      int r = c >> 4, c8 = (c & 15) * 8;
      v8bf v = *(const v8bf*)(
          xv + ((size_t)(b * SEQ + kvBase + r) * NKV + kvh) * HD + c8);
#pragma unroll
      for (int j = 0; j < 8; ++j) Vt[(c8 + j) * LV + r] = v[j];
    }
#ifdef HAVE_TDM
    if (wid == 0) wait_tensorcnt0();                 // Q (first iter) + K done
#endif
    __syncthreads();

    // ---- S = Q K^T : 16 q rows x 64 kv cols per wave ----
    v16bf aq[4];
#pragma unroll
    for (int ks = 0; ks < 4; ++ks)
      aq[ks] = load_frag(&Qs[(wid * 16) * LQ + ks * 32], LQ);
    v8f s[4];
#pragma unroll
    for (int nt = 0; nt < 4; ++nt) {
      s[nt] = vzero;
#pragma unroll
      for (int ks = 0; ks < 4; ++ks) {
        v16bf bk = load_frag(&Ks[(nt * 16) * LQ + ks * 32], LQ);
        s[nt] = wmma_bf16(aq[ks], bk, s[nt]);
      }
    }

    // ---- scale + causal mask + running row max ----
    const int qRow0 = qBase + wid * 16 + hi8;
    float mnew[8];
#pragma unroll
    for (int r = 0; r < 8; ++r) mnew[r] = mrow[r];
#pragma unroll
    for (int nt = 0; nt < 4; ++nt)
#pragma unroll
      for (int r = 0; r < 8; ++r) {
        int col = kvBase + nt * 16 + cN;
        float v = s[nt][r] * scale;
        v = (col <= qRow0 + r) ? v : -1e30f;
        s[nt][r] = v;
        mnew[r] = fmaxf(mnew[r], v);
      }
#pragma unroll
    for (int r = 0; r < 8; ++r)
#pragma unroll
      for (int off = 1; off < 16; off <<= 1)
        mnew[r] = fmaxf(mnew[r], __shfl_xor(mnew[r], off, 32));

    // ---- online softmax update ----
    float corr[8];
#pragma unroll
    for (int r = 0; r < 8; ++r) {
      corr[r] = __expf(mrow[r] - mnew[r]);
      float sum = 0.f;
#pragma unroll
      for (int nt = 0; nt < 4; ++nt) {
        float p = __expf(s[nt][r] - mnew[r]);
        s[nt][r] = p;
        sum += p;
      }
#pragma unroll
      for (int off = 1; off < 16; off <<= 1) sum += __shfl_xor(sum, off, 32);
      lrow[r] = lrow[r] * corr[r] + sum;
      mrow[r] = mnew[r];
    }
#pragma unroll
    for (int dt = 0; dt < 8; ++dt)
#pragma unroll
      for (int r = 0; r < 8; ++r) O[dt][r] *= corr[r];

    // ---- P (C layout) -> per-wave LDS -> A-fragment layout ----
    __bf16* pw = Ps[wid];
#pragma unroll
    for (int nt = 0; nt < 4; ++nt)
#pragma unroll
      for (int r = 0; r < 8; ++r)
        pw[(r + hi8) * LV + nt * 16 + cN] = f2bf(s[nt][r]);
    asm volatile("s_wait_dscnt 0" ::: "memory");     // wave-local LDS RAW

    // ---- O += P V ----
#pragma unroll
    for (int ks = 0; ks < 2; ++ks) {
      v16bf ap = load_frag(&pw[ks * 32], LV);
#pragma unroll
      for (int dt = 0; dt < 8; ++dt) {
        v16bf bv = load_frag(&Vt[(dt * 16) * LV + ks * 32], LV);
        O[dt] = wmma_bf16(ap, bv, O[dt]);
      }
    }
    __syncthreads();                                 // before restaging K/V
  }

  // ---- normalize + store ctx (bf16, [b*S+s, h*HD+d] layout) ----
#pragma unroll
  for (int r = 0; r < 8; ++r) lrow[r] = 1.0f / lrow[r];
#pragma unroll
  for (int dt = 0; dt < 8; ++dt)
#pragma unroll
    for (int r = 0; r < 8; ++r) {
      int row = qBase + wid * 16 + r + hi8;
      int col = dt * 16 + cN;
      ctx[(size_t)(b * SEQ + row) * DIMV + h * HD + col] =
          f2bf(O[dt][r] * lrow[r]);
    }
}

// ---------------------------------------------------------------------------
extern "C" void kernel_launch(void* const* d_in, const int* in_sizes, int n_in,
                              void* d_out, int out_size, void* d_ws,
                              size_t ws_size, hipStream_t stream) {
  (void)in_sizes; (void)n_in; (void)out_size; (void)ws_size;
  const float* x  = (const float*)d_in[0];
  const float* wq = (const float*)d_in[1];
  const float* wk = (const float*)d_in[2];
  const float* wv = (const float*)d_in[3];
  const float* wo = (const float*)d_in[4];
  const float* fc = (const float*)d_in[5];
  const float* fs = (const float*)d_in[6];
  // d_in[7] (mask) and d_in[8] (start_pos): causal mask computed in-kernel.

  const int M = BSZV * SEQ;  // 4096 rows
  __bf16* xq  = (__bf16*)d_ws;                         // 4096 x 4096
  __bf16* xk  = xq + (size_t)M * DIMV;                 // 4096 x 1024
  __bf16* xv  = xk + (size_t)M * NKV * HD;             // 4096 x 1024
  __bf16* ctx = xv + (size_t)M * NKV * HD;             // 4096 x 4096

  dim3 blk(256);
  gemm_wmma_kernel<false, false>
      <<<dim3(DIMV / 128, M / 128), blk, 0, stream>>>(x, wq, xq, M, DIMV, DIMV);
  gemm_wmma_kernel<false, false>
      <<<dim3((NKV * HD) / 128, M / 128), blk, 0, stream>>>(x, wk, xk, M, NKV * HD, DIMV);
  gemm_wmma_kernel<false, false>
      <<<dim3((NKV * HD) / 128, M / 128), blk, 0, stream>>>(x, wv, xv, M, NKV * HD, DIMV);
  rope_kernel<<<(BSZV * SEQ * NH * 64) / 256, blk, 0, stream>>>(xq, fc, fs, NH);
  rope_kernel<<<(BSZV * SEQ * NKV * 64) / 256, blk, 0, stream>>>(xk, fc, fs, NKV);
  attn_kernel<<<dim3(BSZV * NH, SEQ / 128), blk, 0, stream>>>(xq, xk, xv, ctx);
  gemm_wmma_kernel<true, true>
      <<<dim3(DIMV / 128, M / 128), blk, 0, stream>>>(ctx, wo, d_out, M, DIMV, DIMV);
}
